// DeformableAttention_12463995093352
// MI455X (gfx1250) — compile-verified, matrix operand order checked by reading, not run on hardware
//
#include <hip/hip_runtime.h>
#include <hip/hip_bf16.h>

// ---------------- problem constants (from reference) ----------------
#define NG        8192      // gaussians
#define NH        8         // heads
#define HD        64        // head dim
#define AD        512       // attn dims
#define GFD       32        // gaussian feature dim
#define NROW      (NG * 3)  // 24576 token rows
#define KPTS      12        // 6 fixed + 6 learnable sample points
#define IMGC      256
#define IMGW      64
#define IMGH      64
#define NPIX      (IMGW * IMGH)   // 4096
#define OFFIN     (HD + 3)        // 67
#define NLRN      18              // NUM_LEARN * 3
#define SIGCLAMP  9.21f

typedef __attribute__((ext_vector_type(16))) _Float16 v16h;
typedef __attribute__((ext_vector_type(8)))  float    v8f;

__constant__ float FIXS[6][3] = {
    {0.f, 0.f, 0.f}, {1.f, 0.f, 0.f}, {0.f, 1.f, 0.f},
    {0.f, 0.f, 1.f}, {-1.f, 0.f, 0.f}, {0.f, -1.f, 0.f}};

// ---------------- workspace layout (bytes) ----------------
static constexpr size_t OFF_MW = 0;                                       // means_w  [G,3] f32
static constexpr size_t OFF_RT = OFF_MW + (size_t)NG * 3 * 4;             // R^T      [G,9] f32
static constexpr size_t OFF_KM = OFF_RT + (size_t)NG * 9 * 4;             // kmap [512,4096] f32
static constexpr size_t OFF_VM = OFF_KM + (size_t)AD * NPIX * 4;          // vmap [512,4096] f32
static constexpr size_t OFF_QF = OFF_VM + (size_t)AD * NPIX * 4;          // q_full [24576,512] f16
static constexpr size_t OFF_OA = OFF_QF + (size_t)NROW * AD * 2;          // attn out [24576,512] f16

// ---------------- WMMA fragment loaders (ISA 7.12.2 layouts) ----------------
// A: 16x32, row r = lane&15 ; K block = 8*(lane>>4) covering {kb..kb+7, kb+16..kb+23}
template <typename T>
__device__ __forceinline__ v16h load_a_frag(const T* A, long row0, int lda, int k0, int lane) {
    long row = row0 + (lane & 15);
    int  kb  = k0 + ((lane >> 4) << 3);
    const T* p = A + row * (long)lda + kb;
    v16h a;
#pragma unroll
    for (int i = 0; i < 8; ++i) a[i] = (_Float16)(float)p[i];
#pragma unroll
    for (int i = 0; i < 8; ++i) a[8 + i] = (_Float16)(float)p[16 + i];
    return a;
}

// B: 32x16 (row-major [K,N]) ; col = lane&15 ; same K split as A
template <typename T>
__device__ __forceinline__ v16h load_b_frag(const T* B, int k0, long col0, int ldb, int lane) {
    long col = col0 + (lane & 15);
    int  kb  = k0 + ((lane >> 4) << 3);
    const T* p = B + (long)kb * ldb + col;
    v16h b;
#pragma unroll
    for (int i = 0; i < 8; ++i) b[i] = (_Float16)(float)p[(long)i * ldb];
#pragma unroll
    for (int i = 0; i < 8; ++i) b[8 + i] = (_Float16)(float)p[(long)(16 + i) * ldb];
    return b;
}

// ---------------- kernel 1: per-gaussian prep ----------------
__global__ void prep_kernel(const float* __restrict__ means,
                            const float* __restrict__ rotations,
                            const float* __restrict__ transforms,
                            float* __restrict__ meansw,
                            float* __restrict__ rt) {
    int g = blockIdx.x * blockDim.x + threadIdx.x;
    if (g >= NG) return;
    float w = rotations[g * 4 + 0], x = rotations[g * 4 + 1];
    float y = rotations[g * 4 + 2], z = rotations[g * 4 + 3];
    float inv = rsqrtf(w * w + x * x + y * y + z * z);
    w *= inv; x *= inv; y *= inv; z *= inv;
    float R[3][3];
    R[0][0] = 1.f - 2.f * (y * y + z * z); R[0][1] = 2.f * (x * y - w * z); R[0][2] = 2.f * (x * z + w * y);
    R[1][0] = 2.f * (x * y + w * z); R[1][1] = 1.f - 2.f * (x * x + z * z); R[1][2] = 2.f * (y * z - w * x);
    R[2][0] = 2.f * (x * z - w * y); R[2][1] = 2.f * (y * z + w * x); R[2][2] = 1.f - 2.f * (x * x + y * y);
#pragma unroll
    for (int i = 0; i < 3; ++i)
#pragma unroll
        for (int j = 0; j < 3; ++j)
            rt[(long)g * 9 + i * 3 + j] = R[j][i];   // store R^T
    const float* T = transforms + (long)g * 16;
    float m0 = means[g * 3 + 0], m1 = means[g * 3 + 1], m2 = means[g * 3 + 2];
#pragma unroll
    for (int i = 0; i < 3; ++i)
        meansw[g * 3 + i] = T[i * 4 + 0] * m0 + T[i * 4 + 1] * m1 + T[i * 4 + 2] * m2 + T[i * 4 + 3];
}

// ---------------- kernel 2: q projection  [24576,32] x [32,512] ----------------
// one wave -> 16x64 tile (A fragment reused across 4 WMMAs; K = 32 exactly)
__global__ void __launch_bounds__(32)
gemm_qproj_kernel(const float* __restrict__ features, const float* __restrict__ Wq,
                  _Float16* __restrict__ qfull) {
    int  lane = threadIdx.x;
    long row0 = (long)blockIdx.x * 16;
    long col0 = (long)blockIdx.y * 64;
    v16h a = load_a_frag(features, row0, GFD, 0, lane);
    v8f  acc[4];
#pragma unroll
    for (int n = 0; n < 4; ++n) {
        v16h b = load_b_frag(Wq, 0, col0 + n * 16, AD, lane);
        v8f  c = {};
        acc[n] = __builtin_amdgcn_wmma_f32_16x16x32_f16(false, a, false, b, (short)0, c, false, false);
    }
    long rb = row0 + ((lane >> 4) << 3);
    int  cl = lane & 15;
#pragma unroll
    for (int n = 0; n < 4; ++n)
#pragma unroll
        for (int r = 0; r < 8; ++r)
            qfull[(rb + r) * AD + col0 + n * 16 + cl] = (_Float16)acc[n][r];
}

// ---------------- kernel 3: key/value maps  [512,256] x [256,4096] ----------------
// one wave -> 16x64 tile, K-loop of 8; A fragment reused across 4 WMMAs per kk
__global__ void __launch_bounds__(32)
gemm_kv_kernel(const float* __restrict__ Wk, const float* __restrict__ Wv,
               const float* __restrict__ img,
               float* __restrict__ kmap, float* __restrict__ vmap) {
    int lane = threadIdx.x;
    const float* A = blockIdx.z ? Wv : Wk;
    float* D = blockIdx.z ? vmap : kmap;
    long row0 = (long)blockIdx.x * 16;   // channel tile (0..511)
    long col0 = (long)blockIdx.y * 64;   // pixel tile   (0..4095)
    v8f acc[4] = {};
#pragma unroll
    for (int kk = 0; kk < IMGC; kk += 32) {
        if (kk + 32 < IMGC) {
            __builtin_prefetch(A + (row0 + (lane & 15)) * IMGC + kk + 32, 0, 0);
            __builtin_prefetch(img + (long)(kk + 32) * NPIX + col0 + lane, 0, 0);
        }
        v16h a = load_a_frag(A, row0, IMGC, kk, lane);
#pragma unroll
        for (int n = 0; n < 4; ++n) {
            v16h b = load_b_frag(img, kk, col0 + n * 16, NPIX, lane);
            acc[n] = __builtin_amdgcn_wmma_f32_16x16x32_f16(false, a, false, b, (short)0, acc[n],
                                                            false, false);
        }
    }
    long rb = row0 + ((lane >> 4) << 3);
    int  cl = lane & 15;
#pragma unroll
    for (int n = 0; n < 4; ++n)
#pragma unroll
        for (int r = 0; r < 8; ++r)
            D[(rb + r) * NPIX + col0 + n * 16 + cl] = acc[n][r];
}

// ---------------- kernel 4: fused deformable attention ----------------
__device__ __forceinline__ float fetch_px(const float* m, int xi, int yi) {
    bool  v  = (xi >= 0) & (xi < IMGW) & (yi >= 0) & (yi < IMGH);
    int   xc = min(max(xi, 0), IMGW - 1);
    int   yc = min(max(yi, 0), IMGH - 1);
    return v ? m[yc * IMGW + xc] : 0.f;
}

__global__ void __launch_bounds__(32)
fused_attn_kernel(const _Float16* __restrict__ qfull,
                  const float* __restrict__ kmap, const float* __restrict__ vmap,
                  const float* __restrict__ meansw, const float* __restrict__ rt,
                  const float* __restrict__ scales,
                  const float* __restrict__ W_off, const float* __restrict__ b_off,
                  const float* __restrict__ proj,
                  _Float16* __restrict__ outa) {
    int g = blockIdx.x, h = blockIdx.y, tid = threadIdx.x;

    __shared__ float q_s[3][HD];
    __shared__ float offin_s[OFFIN];
    __shared__ float scl_s[KPTS][3];
    __shared__ float sx_s[KPTS], sy_s[KPTS];
    __shared__ float ks_s[KPTS][HD];
    __shared__ float vs_s[KPTS][HD];
    __shared__ float attn_s[3][KPTS];

    long qbase = (long)g * 3 * AD + h * HD;

    // load q and accumulate token-mean
    float qm[2] = {0.f, 0.f};
#pragma unroll
    for (int t = 0; t < 3; ++t) {
#pragma unroll
        for (int j = 0; j < 2; ++j) {
            int   d = tid + j * 32;
            float v = (float)qfull[qbase + (long)t * AD + d];
            q_s[t][d] = v;
            qm[j] += v;
        }
    }
#pragma unroll
    for (int j = 0; j < 2; ++j)
        offin_s[3 + tid + j * 32] = qm[j] * (1.f / 3.f);
    if (tid < 3) offin_s[tid] = meansw[g * 3 + tid];
    __syncthreads();

    // OffsetNet: learnable scales (lanes 0..17) + fixed scales (lanes 18..23)
    if (tid < NLRN) {
        float acc = b_off[tid];
        for (int i = 0; i < OFFIN; ++i) acc += offin_s[i] * W_off[i * NLRN + tid];
        acc = fminf(fmaxf(acc, -SIGCLAMP), SIGCLAMP);
        float sig = 1.f / (1.f + __expf(-acc));
        scl_s[6 + tid / 3][tid % 3] = sig - 0.5f;
    } else if (tid < 24) {
        int k = tid - 18;
        scl_s[k][0] = FIXS[k][0]; scl_s[k][1] = FIXS[k][1]; scl_s[k][2] = FIXS[k][2];
    }
    __syncthreads();

    // rotate, translate, project each sample point (lanes 0..11)
    if (tid < KPTS) {
        float o0 = scl_s[tid][0] * scales[g * 3 + 0];
        float o1 = scl_s[tid][1] * scales[g * 3 + 1];
        float o2 = scl_s[tid][2] * scales[g * 3 + 2];
        const float* R = rt + (long)g * 9;   // R^T
        float p0 = R[0] * o0 + R[1] * o1 + R[2] * o2 + meansw[g * 3 + 0];
        float p1 = R[3] * o0 + R[4] * o1 + R[5] * o2 + meansw[g * 3 + 1];
        float p2 = R[6] * o0 + R[7] * o1 + R[8] * o2 + meansw[g * 3 + 2];
        float px = proj[0] * p0 + proj[1] * p1 + proj[2]  * p2 + proj[3];
        float py = proj[4] * p0 + proj[5] * p1 + proj[6]  * p2 + proj[7];
        float pz = proj[8] * p0 + proj[9] * p1 + proj[10] * p2 + proj[11];
        pz = fmaxf(pz, 1e-5f);
        float nx = fminf(fmaxf(px / pz * (1.f / (float)IMGW), 0.f), 0.9999f);
        float ny = fminf(fmaxf(py / pz * (1.f / (float)IMGH), 0.f), 0.9999f);
        sx_s[tid] = nx * (float)IMGW - 0.5f;   // (grid+1)*W/2 - 0.5
        sy_s[tid] = ny * (float)IMGH - 0.5f;
    }
    __syncthreads();

    // bilinear sampling of k/v maps; lanes cover channels (2 per lane)
    long cb = (long)(h * HD) * NPIX;
    for (int k = 0; k < KPTS; ++k) {
        float x = sx_s[k], y = sy_s[k];
        float x0f = floorf(x), y0f = floorf(y);
        float wx1 = x - x0f, wy1 = y - y0f;
        float wx0 = 1.f - wx1, wy0 = 1.f - wy1;
        int   x0 = (int)x0f, y0 = (int)y0f;
        float w00 = wx0 * wy0, w10 = wx1 * wy0, w01 = wx0 * wy1, w11 = wx1 * wy1;
#pragma unroll
        for (int j = 0; j < 2; ++j) {
            int d = tid + j * 32;
            const float* km = kmap + cb + (long)d * NPIX;
            const float* vm = vmap + cb + (long)d * NPIX;
            ks_s[k][d] = w00 * fetch_px(km, x0, y0)     + w10 * fetch_px(km, x0 + 1, y0)
                       + w01 * fetch_px(km, x0, y0 + 1) + w11 * fetch_px(km, x0 + 1, y0 + 1);
            vs_s[k][d] = w00 * fetch_px(vm, x0, y0)     + w10 * fetch_px(vm, x0 + 1, y0)
                       + w01 * fetch_px(vm, x0, y0 + 1) + w11 * fetch_px(vm, x0 + 1, y0 + 1);
        }
    }
    __syncthreads();

    // scores: 36 (t,k) dot products of length 64
    for (int p = tid; p < 3 * KPTS; p += 32) {
        int t = p / KPTS, k = p % KPTS;
        float s = 0.f;
        for (int d = 0; d < HD; ++d) s += q_s[t][d] * ks_s[k][d];
        attn_s[t][k] = s * 0.125f;    // HD^-0.5
    }
    __syncthreads();

    // softmax over K (lanes 0..2, one token row each)
    if (tid < 3) {
        float m = -1e30f;
        for (int k = 0; k < KPTS; ++k) m = fmaxf(m, attn_s[tid][k]);
        float sum = 0.f;
        for (int k = 0; k < KPTS; ++k) {
            float e = __expf(attn_s[tid][k] - m);
            attn_s[tid][k] = e;
            sum += e;
        }
        float inv = 1.f / sum;
        for (int k = 0; k < KPTS; ++k) attn_s[tid][k] *= inv;
    }
    __syncthreads();

    // out = attn @ V ; store f16 at [g*3+t, h*64+d]
#pragma unroll
    for (int t = 0; t < 3; ++t) {
#pragma unroll
        for (int j = 0; j < 2; ++j) {
            int   d = tid + j * 32;
            float o = 0.f;
            for (int k = 0; k < KPTS; ++k) o += attn_s[t][k] * vs_s[k][d];
            outa[qbase + (long)t * AD + d] = (_Float16)o;
        }
    }
}

// ---------------- kernel 5: output projection + bias + residual ----------------
// one wave -> full 16x32 output rows (A fragment reused across both N tiles)
__global__ void __launch_bounds__(32)
gemm_out_kernel(const _Float16* __restrict__ outa, const float* __restrict__ Wout,
                const float* __restrict__ b_out, const float* __restrict__ features,
                float* __restrict__ out) {
    int  lane = threadIdx.x;
    long row0 = (long)blockIdx.x * 16;
    v8f  acc[2] = {};
#pragma unroll
    for (int kk = 0; kk < AD; kk += 32) {
        if (kk + 32 < AD)
            __builtin_prefetch(outa + (row0 + (lane & 15)) * AD + kk + 32, 0, 0);
        v16h a = load_a_frag(outa, row0, AD, kk, lane);
#pragma unroll
        for (int n = 0; n < 2; ++n) {
            v16h b = load_b_frag(Wout, kk, n * 16, GFD, lane);
            acc[n] = __builtin_amdgcn_wmma_f32_16x16x32_f16(false, a, false, b, (short)0, acc[n],
                                                            false, false);
        }
    }
    long rb = row0 + ((lane >> 4) << 3);
    int  cl = lane & 15;
#pragma unroll
    for (int n = 0; n < 2; ++n) {
        int   col = n * 16 + cl;
        float bo  = b_out[col];
#pragma unroll
        for (int r = 0; r < 8; ++r) {
            long idx = (rb + r) * GFD + col;
            out[idx] = acc[n][r] + bo + features[idx];
        }
    }
}

// ---------------- launcher ----------------
extern "C" void kernel_launch(void* const* d_in, const int* in_sizes, int n_in,
                              void* d_out, int out_size, void* d_ws, size_t ws_size,
                              hipStream_t stream) {
    const float* means      = (const float*)d_in[0];
    const float* scales     = (const float*)d_in[1];
    const float* rotations  = (const float*)d_in[2];
    const float* features   = (const float*)d_in[3];
    const float* transforms = (const float*)d_in[4];
    const float* projection = (const float*)d_in[5];
    const float* img        = (const float*)d_in[6];
    const float* Wq         = (const float*)d_in[7];
    const float* Wk         = (const float*)d_in[8];
    const float* Wv         = (const float*)d_in[9];
    const float* W_off      = (const float*)d_in[10];
    const float* b_off      = (const float*)d_in[11];
    const float* Wout       = (const float*)d_in[12];
    const float* b_out      = (const float*)d_in[13];
    float*       out        = (float*)d_out;

    char* ws = (char*)d_ws;
    float*     meansw = (float*)(ws + OFF_MW);
    float*     rt     = (float*)(ws + OFF_RT);
    float*     kmap   = (float*)(ws + OFF_KM);
    float*     vmap   = (float*)(ws + OFF_VM);
    _Float16*  qfull  = (_Float16*)(ws + OFF_QF);
    _Float16*  outa   = (_Float16*)(ws + OFF_OA);

    prep_kernel<<<dim3(NG / 256), dim3(256), 0, stream>>>(means, rotations, transforms, meansw, rt);

    gemm_qproj_kernel<<<dim3(NROW / 16, AD / 64), dim3(32), 0, stream>>>(features, Wq, qfull);

    gemm_kv_kernel<<<dim3(AD / 16, NPIX / 64, 2), dim3(32), 0, stream>>>(Wk, Wv, img, kmap, vmap);

    fused_attn_kernel<<<dim3(NG, NH), dim3(32), 0, stream>>>(
        qfull, kmap, vmap, meansw, rt, scales, W_off, b_off, projection, outa);

    gemm_out_kernel<<<dim3(NROW / 16), dim3(32), 0, stream>>>(outa, Wout, b_out, features, out);
}
